// Block_71004399338153
// MI455X (gfx1250) — matrix operand order repaired and checked by reading, not compiled, not run on hardware
//
#include <hip/hip_runtime.h>
#include <math.h>
#include <stdint.h>

#define HEADSZ 64
#define CDIM   1024
#define FDIM   4096
#define BB     8
#define TT     256
#define HH     (CDIM / HEADSZ)   // 16
#define BT     (BB * TT)         // 2048
#define GN_EPS 6.4e-4f           // 1e-5 * 8^2

typedef __attribute__((ext_vector_type(16))) __bf16 v16bf;
typedef __attribute__((ext_vector_type(8)))  __bf16 v8bf;
typedef __attribute__((ext_vector_type(4)))  __bf16 v4bf;
typedef __attribute__((ext_vector_type(8)))  float  v8f;
typedef __attribute__((ext_vector_type(4)))  float  f4;

// ---------------------------------------------------------------------------
// LayerNorm over C per row (BT blocks x 256 threads, float4 per thread)
// ---------------------------------------------------------------------------
__global__ void k_layernorm(const float* __restrict__ x, const float* __restrict__ g,
                            const float* __restrict__ b, float* __restrict__ out) {
  int row = blockIdx.x;
  int tid = threadIdx.x;
  const f4* xr4 = (const f4*)(x + (size_t)row * CDIM);
  f4 v = xr4[tid];
  float ssum = v[0] + v[1] + v[2] + v[3];
  float ssq  = v[0]*v[0] + v[1]*v[1] + v[2]*v[2] + v[3]*v[3];
  __shared__ float s1[256], s2[256];
  s1[tid] = ssum; s2[tid] = ssq; __syncthreads();
  for (int off = 128; off > 0; off >>= 1) {
    if (tid < off) { s1[tid] += s1[tid + off]; s2[tid] += s2[tid + off]; }
    __syncthreads();
  }
  float mean = s1[0] * (1.f / CDIM);
  float var  = s2[0] * (1.f / CDIM) - mean * mean;
  float rs   = rsqrtf(var + 1e-5f);
  f4 o;
#pragma unroll
  for (int i = 0; i < 4; ++i) {
    int c = tid * 4 + i;
    o[i] = (v[i] - mean) * rs * g[c] + b[c];
  }
  ((f4*)(out + (size_t)row * CDIM))[tid] = o;
}

// ---------------------------------------------------------------------------
// xx = bshift(xa) - xa   (per-head: low 32 chans from t-1, high 32 from t+1)
// ---------------------------------------------------------------------------
__global__ void k_bshift_delta(const float* __restrict__ xa, float* __restrict__ xx) {
  size_t idx = (size_t)blockIdx.x * 256 + threadIdx.x;
  int    c   = (int)(idx % CDIM);
  size_t bt  = idx / CDIM;
  int    t   = (int)(bt % TT);
  int    hc  = c & (HEADSZ - 1);
  float  src = 0.f;
  if (hc < HEADSZ / 2) { if (t > 0)      src = xa[(bt - 1) * CDIM + c]; }
  else                 { if (t < TT - 1) src = xa[(bt + 1) * CDIM + c]; }
  xx[idx] = src - xa[idx];
}

// ---------------------------------------------------------------------------
// a = tanh((xa + xx*maa_x) @ maa_w1)   [BT x 160]
// ---------------------------------------------------------------------------
__global__ void k_mix_a(const float* __restrict__ xa, const float* __restrict__ xx,
                        const float* __restrict__ maa_x, const float* __restrict__ w1,
                        float* __restrict__ a) {
  int bt = blockIdx.x, tid = threadIdx.x;   // 160 threads
  __shared__ float mix[CDIM];
  size_t base = (size_t)bt * CDIM;
  for (int c = tid; c < CDIM; c += 160)
    mix[c] = xa[base + c] + xx[base + c] * maa_x[c];
  __syncthreads();
  float acc = 0.f;
  for (int c = 0; c < CDIM; ++c) acc += mix[c] * w1[(size_t)c * 160 + tid];
  a[(size_t)bt * 160 + tid] = tanhf(acc);
}

// ---------------------------------------------------------------------------
// m_f = a_f @ maa_w2[f]; x{w,k,v,r,g} = xa + xx*(maa_* + m_f)
// ---------------------------------------------------------------------------
__global__ void k_mix5(const float* __restrict__ xa, const float* __restrict__ xx,
                       const float* __restrict__ a,  const float* __restrict__ w2,
                       const float* __restrict__ mw, const float* __restrict__ mk,
                       const float* __restrict__ mv, const float* __restrict__ mr,
                       const float* __restrict__ mg,
                       float* __restrict__ xw, float* __restrict__ xk,
                       float* __restrict__ xv, float* __restrict__ xr,
                       float* __restrict__ xg) {
  int bt = blockIdx.x, tid = threadIdx.x;   // 256 threads
  __shared__ float as[160];
  if (tid < 160) as[tid] = a[(size_t)bt * 160 + tid];
  __syncthreads();
  size_t base = (size_t)bt * CDIM;
  for (int c = tid; c < CDIM; c += 256) {
    float xav = xa[base + c], xxv = xx[base + c];
    float m[5];
#pragma unroll
    for (int f = 0; f < 5; ++f) {
      float s = 0.f;
#pragma unroll
      for (int d = 0; d < 32; ++d) s += as[f * 32 + d] * w2[(size_t)(f * 32 + d) * CDIM + c];
      m[f] = s;
    }
    xw[base + c] = xav + xxv * (mw[c] + m[0]);
    xk[base + c] = xav + xxv * (mk[c] + m[1]);
    xv[base + c] = xav + xxv * (mv[c] + m[2]);
    xr[base + c] = xav + xxv * (mr[c] + m[3]);
    xg[base + c] = xav + xxv * (mg[c] + m[4]);
  }
}

// ---------------------------------------------------------------------------
// ww = tanh(xw @ tdw1)  [BT x 64]
// ---------------------------------------------------------------------------
__global__ void k_tdw1(const float* __restrict__ xw, const float* __restrict__ w1,
                       float* __restrict__ ww) {
  int bt = blockIdx.x, tid = threadIdx.x;   // 64 threads
  __shared__ float s[CDIM];
  size_t base = (size_t)bt * CDIM;
  for (int c = tid; c < CDIM; c += 64) s[c] = xw[base + c];
  __syncthreads();
  float acc = 0.f;
  for (int c = 0; c < CDIM; ++c) acc += s[c] * w1[(size_t)c * 64 + tid];
  ww[(size_t)bt * 64 + tid] = tanhf(acc);
}

// ---------------------------------------------------------------------------
// dec = exp(-exp(time_decay + ww @ tdw2))  [BT x C]
// ---------------------------------------------------------------------------
__global__ void k_decay(const float* __restrict__ ww, const float* __restrict__ w2,
                        const float* __restrict__ td, float* __restrict__ dec) {
  int bt = blockIdx.x, tid = threadIdx.x;   // 256 threads
  __shared__ float s[64];
  if (tid < 64) s[tid] = ww[(size_t)bt * 64 + tid];
  __syncthreads();
  size_t base = (size_t)bt * CDIM;
  for (int c = tid; c < CDIM; c += 256) {
    float acc = td[c];
#pragma unroll
    for (int j = 0; j < 64; ++j) acc += s[j] * w2[(size_t)j * CDIM + c];
    dec[base + c] = __expf(-__expf(acc));
  }
}

// ---------------------------------------------------------------------------
// bf16 WMMA GEMM (templated N,K so B-tile loads use immediate offsets):
// Out[M,N] = epi(A[M,K] @ B[K,N]) (+ res)
// Block tile 32x128, 8 waves (2x4), wave tile 16x32 (2 accumulators sharing
// the A fragment). LDS ping-pong double buffering: one barrier per K-step,
// next tile's global loads overlap current tile's WMMAs.
// A staged row-major bf16, B staged K-major (transposed) so fragments are
// contiguous 2x ds_load_b128 per lane. epi: 0=none,1=relu^2,2=silu,3=sigmoid.
// ---------------------------------------------------------------------------
#define LDA 40   // bf16 stride: 80 bytes, 16B-aligned, conflict-breaking
#define LDB 40

template<int N, int K>
__global__ void k_gemm_bf16(const float* __restrict__ A, const float* __restrict__ Bm,
                            const float* __restrict__ res, float* __restrict__ Out,
                            int M, int epi) {
  __shared__ __bf16 As[2][32][LDA];     // [buf][row][k]
  __shared__ __bf16 Bs[2][128][LDB];    // [buf][col][k]  (transposed tile)
  int tid  = threadIdx.x;
  int wave = tid >> 5, lane = tid & 31;
  int wm = wave >> 2, wn = wave & 3;        // wave tile: rows wm*16, cols wn*32
  int lh = lane >> 4, l15 = lane & 15;
  int m0 = blockIdx.y * 32, n0 = blockIdx.x * 128;

  int ar = tid & 31,  ak = (tid >> 5) << 2;   // A: 1 row, 4 consecutive k
  int bc = tid & 127, bk = (tid >> 7) << 4;   // B: 1 col, 16 consecutive k

  const float* arow_p = &A[(size_t)(m0 + ar) * K + ak];
  const float* bcol_p = &Bm[(size_t)bk * N + n0 + bc];

  f4    areg;
  float breg[16];
  auto gload = [&](int k0) {
    areg = *(const f4*)(arow_p + k0);
    const float* bp = bcol_p + (size_t)k0 * N;
#pragma unroll
    for (int j = 0; j < 16; ++j) breg[j] = bp[(size_t)j * N];  // imm offsets (N constexpr)
  };
  auto lstore = [&](int buf) {
    v4bf pa;
#pragma unroll
    for (int j = 0; j < 4; ++j) pa[j] = (__bf16)areg[j];
    *(v4bf*)&As[buf][ar][ak] = pa;
    v8bf p0, p1;
#pragma unroll
    for (int j = 0; j < 8; ++j) { p0[j] = (__bf16)breg[j]; p1[j] = (__bf16)breg[8 + j]; }
    *(v8bf*)&Bs[buf][bc][bk]     = p0;
    *(v8bf*)&Bs[buf][bc][bk + 8] = p1;
  };

  v8f acc0 = {}, acc1 = {};

  gload(0);
  lstore(0);
  __syncthreads();

  for (int k0 = 0; k0 < K; k0 += 32) {
    int cur = (k0 >> 5) & 1;
    if (k0 + 32 < K) gload(k0 + 32);            // overlap with WMMAs below
    if (k0 + 64 < K) {                          // global_prefetch tile i+2
      __builtin_prefetch(arow_p + k0 + 64, 0, 1);
      __builtin_prefetch(bcol_p + (size_t)(k0 + 64) * N, 0, 1);
    }
    // ---- fragments: contiguous 16B LDS reads, k = 16*half + s for A and B
    int arow = wm * 16 + l15;
    v8bf alo = *(const v8bf*)&As[cur][arow][lh * 16];
    v8bf ahi = *(const v8bf*)&As[cur][arow][lh * 16 + 8];
    v16bf afr = __builtin_shufflevector(alo, ahi,
        0, 1, 2, 3, 4, 5, 6, 7, 8, 9, 10, 11, 12, 13, 14, 15);
    int c0 = wn * 32 + l15;
    v8bf b0l = *(const v8bf*)&Bs[cur][c0][lh * 16];
    v8bf b0h = *(const v8bf*)&Bs[cur][c0][lh * 16 + 8];
    v16bf bf0 = __builtin_shufflevector(b0l, b0h,
        0, 1, 2, 3, 4, 5, 6, 7, 8, 9, 10, 11, 12, 13, 14, 15);
    int c1 = c0 + 16;
    v8bf b1l = *(const v8bf*)&Bs[cur][c1][lh * 16];
    v8bf b1h = *(const v8bf*)&Bs[cur][c1][lh * 16 + 8];
    v16bf bf1 = __builtin_shufflevector(b1l, b1h,
        0, 1, 2, 3, 4, 5, 6, 7, 8, 9, 10, 11, 12, 13, 14, 15);

    acc0 = __builtin_amdgcn_wmma_f32_16x16x32_bf16(false, afr, false, bf0,
                                                   (short)0, acc0, false, false);
    acc1 = __builtin_amdgcn_wmma_f32_16x16x32_bf16(false, afr, false, bf1,
                                                   (short)0, acc1, false, false);

    if (k0 + 32 < K) lstore(cur ^ 1);           // write other buffer, no race
    __syncthreads();                            // single barrier per K-step
  }

#pragma unroll
  for (int t = 0; t < 2; ++t) {
    v8f acc = t ? acc1 : acc0;
#pragma unroll
    for (int i = 0; i < 8; ++i) {
      int rr = m0 + wm * 16 + i + 8 * lh;    // C/D layout: VGPR i -> M = i + 8*half
      int cc = n0 + wn * 32 + t * 16 + l15;
      size_t o = (size_t)rr * N + cc;
      float val = acc[i];
      if (epi == 1)      { float u = fmaxf(val, 0.f); val = u * u; }
      else if (epi == 2) { val = val / (1.f + __expf(-val)); }
      else if (epi == 3) { val = 1.f / (1.f + __expf(-val)); }
      if (res) val += res[o];
      Out[o] = val;
    }
  }
}

// ---------------------------------------------------------------------------
// Bidirectional WKV6 scan. One block per (b,h), 64 threads; thread i owns
// state column S[:,i] (64 f32 in registers). r/k/d broadcast rows are copied
// global->LDS with CDNA5 async-to-LDS ops (ASYNCcnt), no VGPR bounce.
// ---------------------------------------------------------------------------
__device__ __forceinline__ void async_row_to_lds(float* lds_dst, const float* gsrc) {
  unsigned loff = (unsigned)(uintptr_t)lds_dst;   // low 32 bits = LDS offset
  asm volatile("global_load_async_to_lds_b32 %0, %1, off"
               :: "v"(loff), "v"(gsrc) : "memory");
}
__device__ __forceinline__ void wait_async_lds() {
  asm volatile("s_wait_asynccnt 0" ::: "memory");
}

__global__ void k_wkv(const float* __restrict__ r, const float* __restrict__ k,
                      const float* __restrict__ v, const float* __restrict__ dec,
                      const float* __restrict__ u, float* __restrict__ y) {
  int bh = blockIdx.x, b = bh / HH, h = bh % HH;
  int i = threadIdx.x;
  __shared__ float rs[64], ks[64], ds[64], us[64];
  us[i] = u[h * HEADSZ + i];
  float S[64];
#pragma unroll
  for (int j = 0; j < 64; ++j) S[j] = 0.f;

  // forward: y = r . (S + u*k v^T), then S = d*S + k v^T
  for (int t = 0; t < TT; ++t) {
    size_t base = (size_t)(b * TT + t) * CDIM + h * HEADSZ;
    async_row_to_lds(&rs[i], &r[base + i]);
    async_row_to_lds(&ks[i], &k[base + i]);
    async_row_to_lds(&ds[i], &dec[base + i]);
    float vi = v[base + i];
    wait_async_lds();
    __syncthreads();
    float acc = 0.f;
#pragma unroll
    for (int j = 0; j < 64; ++j) {
      float kv = ks[j] * vi;
      acc += rs[j] * (S[j] + us[j] * kv);
      S[j] = ds[j] * S[j] + kv;
    }
    y[base + i] = acc;
    __syncthreads();
  }

#pragma unroll
  for (int j = 0; j < 64; ++j) S[j] = 0.f;

  // backward: y += r . S (pre-update), then S = d*S + k v^T
  for (int t = TT - 1; t >= 0; --t) {
    size_t base = (size_t)(b * TT + t) * CDIM + h * HEADSZ;
    async_row_to_lds(&rs[i], &r[base + i]);
    async_row_to_lds(&ks[i], &k[base + i]);
    async_row_to_lds(&ds[i], &dec[base + i]);
    float vi = v[base + i];
    wait_async_lds();
    __syncthreads();
    float acc = 0.f;
#pragma unroll
    for (int j = 0; j < 64; ++j) acc += rs[j] * S[j];
#pragma unroll
    for (int j = 0; j < 64; ++j) S[j] = ds[j] * S[j] + ks[j] * vi;
    y[base + i] += acc;
    __syncthreads();
  }
}

// ---------------------------------------------------------------------------
// GroupNorm(64-chan heads) then multiply by g. One block per (bt, h).
// ---------------------------------------------------------------------------
__global__ void k_gn_mulg(const float* __restrict__ y, const float* __restrict__ g,
                          const float* __restrict__ gw, const float* __restrict__ gb,
                          float* __restrict__ out) {
  int bth = blockIdx.x, bt = bth / HH, h = bth % HH;
  int i = threadIdx.x;
  size_t base = (size_t)bt * CDIM + h * HEADSZ;
  float v = y[base + i];
  __shared__ float s1[64], s2[64];
  s1[i] = v; s2[i] = v * v; __syncthreads();
  for (int off = 32; off > 0; off >>= 1) {
    if (i < off) { s1[i] += s1[i + off]; s2[i] += s2[i + off]; }
    __syncthreads();
  }
  float m   = s1[0] * (1.f / HEADSZ);
  float var = s2[0] * (1.f / HEADSZ) - m * m;
  float nv  = (v - m) * rsqrtf(var + GN_EPS);
  int c = h * HEADSZ + i;
  out[base + i] = (nv * gw[c] + gb[c]) * g[base + i];
}

// ---------------------------------------------------------------------------
// ckx/crx = xc + (bshift(xc)-xc) * cmaa_{k,r}
// ---------------------------------------------------------------------------
__global__ void k_cmix(const float* __restrict__ xc, const float* __restrict__ cmk,
                       const float* __restrict__ cmr, float* __restrict__ ckx,
                       float* __restrict__ crx) {
  size_t idx = (size_t)blockIdx.x * 256 + threadIdx.x;
  int    c   = (int)(idx % CDIM);
  size_t bt  = idx / CDIM;
  int    t   = (int)(bt % TT);
  int    hc  = c & (HEADSZ - 1);
  float  src = 0.f;
  if (hc < HEADSZ / 2) { if (t > 0)      src = xc[(bt - 1) * CDIM + c]; }
  else                 { if (t < TT - 1) src = xc[(bt + 1) * CDIM + c]; }
  float xcv = xc[idx];
  float xxv = src - xcv;
  ckx[idx] = xcv + xxv * cmk[c];
  crx[idx] = xcv + xxv * cmr[c];
}

// ---------------------------------------------------------------------------
// out = x2 + sigmoid_rr * kv
// ---------------------------------------------------------------------------
__global__ void k_final(const float* __restrict__ x2, const float* __restrict__ sr,
                        const float* __restrict__ kvb, float* __restrict__ out) {
  size_t idx = (size_t)blockIdx.x * 256 + threadIdx.x;
  out[idx] = x2[idx] + sr[idx] * kvb[idx];
}

// ---------------------------------------------------------------------------
extern "C" void kernel_launch(void* const* d_in, const int* in_sizes, int n_in,
                              void* d_out, int out_size, void* d_ws, size_t ws_size,
                              hipStream_t stream) {
  (void)in_sizes; (void)n_in; (void)out_size; (void)ws_size;
  const float* x      = (const float*)d_in[0];
  const float* ln1_g  = (const float*)d_in[1];
  const float* ln1_b  = (const float*)d_in[2];
  const float* ln2_g  = (const float*)d_in[3];
  const float* ln2_b  = (const float*)d_in[4];
  const float* maa_x  = (const float*)d_in[5];
  const float* maa_w  = (const float*)d_in[6];
  const float* maa_k  = (const float*)d_in[7];
  const float* maa_v  = (const float*)d_in[8];
  const float* maa_r  = (const float*)d_in[9];
  const float* maa_g  = (const float*)d_in[10];
  const float* maa_w1 = (const float*)d_in[11];
  const float* maa_w2 = (const float*)d_in[12];
  const float* tdecay = (const float*)d_in[13];
  const float* tdw1   = (const float*)d_in[14];
  const float* tdw2   = (const float*)d_in[15];
  const float* u      = (const float*)d_in[16];
  const float* Wr     = (const float*)d_in[17];
  const float* Wk     = (const float*)d_in[18];
  const float* Wv     = (const float*)d_in[19];
  const float* Wg     = (const float*)d_in[20];
  const float* Wo     = (const float*)d_in[21];
  const float* lnx_g  = (const float*)d_in[22];
  const float* lnx_b  = (const float*)d_in[23];
  const float* cmaa_k = (const float*)d_in[24];
  const float* cmaa_r = (const float*)d_in[25];
  const float* Wck    = (const float*)d_in[26];
  const float* Wcv    = (const float*)d_in[27];
  const float* Wcr    = (const float*)d_in[28];
  float* out = (float*)d_out;

  const size_t NBTC = (size_t)BT * CDIM;
  float* p = (float*)d_ws;
  auto take = [&](size_t n) { float* q = p; p += n; return q; };
  float* xa   = take(NBTC);
  float* xx   = take(NBTC);
  float* xw   = take(NBTC);
  float* xk   = take(NBTC);
  float* xv   = take(NBTC);
  float* xr   = take(NBTC);
  float* xg   = take(NBTC);
  float* rb   = take(NBTC);
  float* kb   = take(NBTC);
  float* vb   = take(NBTC);
  float* dec  = take(NBTC);
  float* gb   = take(NBTC);
  float* yb   = take(NBTC);
  float* gn   = take(NBTC);
  float* x2   = take(NBTC);
  float* xc   = take(NBTC);
  float* ckx  = take(NBTC);
  float* crx  = take(NBTC);
  float* kvb  = take(NBTC);
  float* rrs  = take(NBTC);
  float* a160 = take((size_t)BT * 160);
  float* ww64 = take((size_t)BT * 64);
  float* kkb  = take((size_t)BT * FDIM);

  const int EW_BLOCKS = (int)(NBTC / 256);

  // ---- attention branch ----
  k_layernorm<<<BT, 256, 0, stream>>>(x, ln1_g, ln1_b, xa);
  k_bshift_delta<<<EW_BLOCKS, 256, 0, stream>>>(xa, xx);
  k_mix_a<<<BT, 160, 0, stream>>>(xa, xx, maa_x, maa_w1, a160);
  k_mix5<<<BT, 256, 0, stream>>>(xa, xx, a160, maa_w2,
                                 maa_w, maa_k, maa_v, maa_r, maa_g,
                                 xw, xk, xv, xr, xg);
  k_tdw1<<<BT, 64, 0, stream>>>(xw, tdw1, ww64);
  k_decay<<<BT, 256, 0, stream>>>(ww64, tdw2, tdecay, dec);

  dim3 gCC(CDIM / 128, BT / 32);
  k_gemm_bf16<CDIM, CDIM><<<gCC, 256, 0, stream>>>(xr, Wr, nullptr, rb, BT, 0);
  k_gemm_bf16<CDIM, CDIM><<<gCC, 256, 0, stream>>>(xk, Wk, nullptr, kb, BT, 0);
  k_gemm_bf16<CDIM, CDIM><<<gCC, 256, 0, stream>>>(xv, Wv, nullptr, vb, BT, 0);
  k_gemm_bf16<CDIM, CDIM><<<gCC, 256, 0, stream>>>(xg, Wg, nullptr, gb, BT, 2); // silu

  k_wkv<<<BB * HH, 64, 0, stream>>>(rb, kb, vb, dec, u, yb);
  k_gn_mulg<<<BT * HH, 64, 0, stream>>>(yb, gb, lnx_g, lnx_b, gn);
  k_gemm_bf16<CDIM, CDIM><<<gCC, 256, 0, stream>>>(gn, Wo, x, x2, BT, 0); // + residual

  // ---- FFN branch ----
  k_layernorm<<<BT, 256, 0, stream>>>(x2, ln2_g, ln2_b, xc);
  k_cmix<<<EW_BLOCKS, 256, 0, stream>>>(xc, cmaa_k, cmaa_r, ckx, crx);

  dim3 gCF(FDIM / 128, BT / 32);
  k_gemm_bf16<FDIM, CDIM><<<gCF, 256, 0, stream>>>(ckx, Wck, nullptr, kkb, BT, 1); // relu^2
  k_gemm_bf16<CDIM, FDIM><<<gCC, 256, 0, stream>>>(kkb, Wcv, nullptr, kvb, BT, 0);
  k_gemm_bf16<CDIM, CDIM><<<gCC, 256, 0, stream>>>(crx, Wcr, nullptr, rrs, BT, 3); // sigmoid

  k_final<<<EW_BLOCKS, 256, 0, stream>>>(x2, rrs, kvb, out);
}
